// RecurrentAE_30932354465868
// MI455X (gfx1250) — matrix operand order repaired and checked by reading, not compiled, not run on hardware
//
#include <hip/hip_runtime.h>
#include <hip/hip_bf16.h>

// ---------------------------------------------------------------------------
// RecurrentAE on MI455X (gfx1250), persistent-kernel version.
//   - GRU encoder (512 steps) + GRUCell decoder (512 steps) + Linear output.
//   - All GEMMs on v_wmma_f32_16x16x32_bf16 (f32 accumulate).
//   - Weights pre-packed once into WMMA B-fragment lane order (bf16), then
//     pinned in LDS for the whole recurrence (zero global weight traffic in
//     the steady state; encoder->decoder swaps the LDS image once).
//   - Single kernel for all 1024 recurrent steps; device-wide barrier via a
//     monotonic agent-scope atomic counter + s_sleep spin.
//   - Ping-pong state selected by base + cur*size offsets (NOT pointer
//     selects) so h loads stay GLOBAL, not FLAT.
// ---------------------------------------------------------------------------

typedef __attribute__((ext_vector_type(16))) __bf16 v16bf;
typedef __attribute__((ext_vector_type(8)))  __bf16 v8bf;
typedef __attribute__((ext_vector_type(8)))  float  v8f;

#define B_ 128
#define S_ 512
#define F_ 128
#define L_ 1024

#define KBH (L_ / 32)          // 32 K-blocks for hidden projection
#define KBI (F_ / 32)          // 4  K-blocks for input projection
#define KBO (L_ / 32)          // 32 K-blocks for output projection
#define NTG (L_ / 16)          // 64 latent column tiles per gate
#define GRID_ 64               // one block per latent column tile

#define HSTRIDE ((size_t)B_ * L_)   // one h buffer (elements)
#define XSTRIDE ((size_t)B_ * F_)   // one decoder-input buffer (elements)

// LDS image: per-block weight slices in B-fragment order.
#define WHH_OFF   0
#define WHH_BYTES (3 * KBH * 32 * 32)          // 98304
#define WIH_OFF   (WHH_OFF + WHH_BYTES)
#define WIH_BYTES (3 * KBI * 32 * 32)          // 12288
#define WOUT_OFF  (WIH_OFF + WIH_BYTES)
#define WOUT_BYTES (KBO * 32 * 32)             // 32768
#define SMEM_BYTES (WOUT_OFF + WOUT_BYTES)     // 143360 (<= 320KB WGP LDS)

__device__ __forceinline__ float sigmoidf_(float x) {
    return 1.0f / (1.0f + __expf(-x));
}
__device__ __forceinline__ float tanhf_(float x) {
    float e = __expf(-2.0f * fabsf(x));
    float t = (1.0f - e) / (1.0f + e);
    return copysignf(t, x);
}

// A-fragment (16x32 bf16, MxK) per ISA layout: row = lane&15, khalf = lane>>4
// selects K in {kb..kb+7, kb+16..kb+23} (khalf=0) or {kb+8..15, kb+24..31}.
__device__ __forceinline__ v16bf load_a_frag(const __bf16* __restrict__ rowp,
                                             int kbase, int khalf) {
    const v8bf l0 = *reinterpret_cast<const v8bf*>(rowp + kbase + khalf * 8);
    const v8bf l1 = *reinterpret_cast<const v8bf*>(rowp + kbase + 16 + khalf * 8);
    return __builtin_shufflevector(l0, l1, 0, 1, 2, 3, 4, 5, 6, 7,
                                           8, 9, 10, 11, 12, 13, 14, 15);
}

// B-fragment from LDS: lane reads its 32 contiguous bytes.
__device__ __forceinline__ v16bf lds_frag(const char* smem, int byte_off) {
    return *reinterpret_cast<const v16bf*>(smem + byte_off);
}

__device__ __forceinline__ void copy_to_lds(char* smem, int dst_off,
                                            const __bf16* __restrict__ src,
                                            int nbytes) {
    const uint4* s = reinterpret_cast<const uint4*>(src);
    uint4* d = reinterpret_cast<uint4*>(smem + dst_off);
    const int n = nbytes >> 4;
    for (int i = threadIdx.x; i < n; i += 256) d[i] = s[i];
}

// Device-wide barrier: monotonic counter, agent scope, s_sleep spin.
__device__ __forceinline__ void grid_barrier(unsigned* __restrict__ bar,
                                             unsigned epoch) {
    __threadfence();           // flush this thread's stores to device scope
    __syncthreads();
    if (threadIdx.x == 0) {
        __hip_atomic_fetch_add(bar, 1u, __ATOMIC_RELEASE,
                               __HIP_MEMORY_SCOPE_AGENT);
        const unsigned target = epoch * (unsigned)GRID_;
        while (__hip_atomic_load(bar, __ATOMIC_RELAXED,
                                 __HIP_MEMORY_SCOPE_AGENT) < target)
            __builtin_amdgcn_s_sleep(2);
        (void)__hip_atomic_load(bar, __ATOMIC_ACQUIRE,
                                __HIP_MEMORY_SCOPE_AGENT);  // invalidate WGP$
    }
    __syncthreads();
}

// ---------------------------------------------------------------------------
// One-time prep kernels
// ---------------------------------------------------------------------------

// Pack row-major f32 W [N,K] (out[n] = sum_k in[k]*W[n][k]) into bf16 WMMA
// B-fragment order: tid = ((ntile*KB + kblock)*32 + lane)*8 + v,
// n = lane&15, khalf = lane>>4, k = kblock*32 + (v&4)*4 + khalf*8 + 2*(v&3) + {0,1}.
__global__ void prepack_kernel(const float* __restrict__ W,
                               __bf16* __restrict__ dst, int N, int K) {
    int tid = blockIdx.x * blockDim.x + threadIdx.x;
    int total = (N * K) >> 1;
    if (tid >= total) return;
    int v     = tid & 7;
    int lane  = (tid >> 3) & 31;
    int KB    = K >> 5;
    int kblk  = (tid >> 8) % KB;
    int ntile = (tid >> 8) / KB;
    int n     = lane & 15;
    int khalf = lane >> 4;
    int k0    = kblk * 32 + (v & 4) * 4 + khalf * 8 + 2 * (v & 3);
    int o     = ntile * 16 + n;
    dst[(size_t)tid * 2 + 0] = (__bf16)W[(size_t)o * K + k0];
    dst[(size_t)tid * 2 + 1] = (__bf16)W[(size_t)o * K + k0 + 1];
}

// x [B,S,F] f32 -> xT [S,B,F] bf16 (time-major slices for the scan).
__global__ void xconv_kernel(const float* __restrict__ x,
                             __bf16* __restrict__ xT) {
    int tid = blockIdx.x * blockDim.x + threadIdx.x;
    if (tid >= B_ * S_ * F_) return;
    int f = tid % F_;
    int t = (tid / F_) % S_;
    int b = tid / (F_ * S_);
    xT[((size_t)t * B_ + b) * F_ + f] = (__bf16)x[tid];
}

// Zero h0 (f32 + bf16 shadow, buffer 0) and the barrier counter.
__global__ void init_kernel(float* __restrict__ hf, __bf16* __restrict__ hb,
                            unsigned* __restrict__ bar) {
    int tid = blockIdx.x * blockDim.x + threadIdx.x;
    if (tid < B_ * L_) {
        hf[tid] = 0.0f;
        hb[tid] = (__bf16)0.0f;
    }
    if (tid == 0) *bar = 0u;
}

// ---------------------------------------------------------------------------
// GRU step (device): one block handles 16 latent columns, wave = M tile.
// Accs: r and z merged (ih+hh), n kept split (r scales only the hh part).
// All global pointers are base + uniform offsets of kernel args -> global ISA.
// ---------------------------------------------------------------------------
__device__ __forceinline__ void gru_step_dev(
    const char* smem,
    const __bf16* __restrict__ hb_base, const float* __restrict__ hf_base,
    size_t off_in, size_t off_out,
    const __bf16* __restrict__ xb,
    float br, float bz, float bni, float bnh,
    float* __restrict__ hf_base_w, __bf16* __restrict__ hb_base_w,
    int blk, int lane, int wave)
{
    const int khalf = lane >> 4;
    const int mloc  = lane & 15;
    const int arow  = wave * 16 + mloc;
    const int col   = blk * 16 + mloc;

    v8f ar = {}, az = {}, ani = {}, anh = {};

    // input projection: K = F = 128 (weights from LDS)
    const __bf16* xrow = xb + (size_t)arow * F_;
    #pragma unroll
    for (int kb = 0; kb < KBI; ++kb) {
        v16bf a  = load_a_frag(xrow, kb * 32, khalf);
        v16bf b0 = lds_frag(smem, WIH_OFF + (((0 * KBI + kb) * 32 + lane) << 5));
        v16bf b1 = lds_frag(smem, WIH_OFF + (((1 * KBI + kb) * 32 + lane) << 5));
        v16bf b2 = lds_frag(smem, WIH_OFF + (((2 * KBI + kb) * 32 + lane) << 5));
        ar  = __builtin_amdgcn_wmma_f32_16x16x32_bf16(false, a, false, b0, (short)0, ar,  false, false);
        az  = __builtin_amdgcn_wmma_f32_16x16x32_bf16(false, a, false, b1, (short)0, az,  false, false);
        ani = __builtin_amdgcn_wmma_f32_16x16x32_bf16(false, a, false, b2, (short)0, ani, false, false);
    }

    // hidden projection: K = L = 1024 (weights from LDS)
    const __bf16* hrow = hb_base + off_in + (size_t)arow * L_;
    #pragma unroll 2
    for (int kb = 0; kb < KBH; ++kb) {
        v16bf a  = load_a_frag(hrow, kb * 32, khalf);
        v16bf b0 = lds_frag(smem, WHH_OFF + (((0 * KBH + kb) * 32 + lane) << 5));
        v16bf b1 = lds_frag(smem, WHH_OFF + (((1 * KBH + kb) * 32 + lane) << 5));
        v16bf b2 = lds_frag(smem, WHH_OFF + (((2 * KBH + kb) * 32 + lane) << 5));
        ar  = __builtin_amdgcn_wmma_f32_16x16x32_bf16(false, a, false, b0, (short)0, ar,  false, false);
        az  = __builtin_amdgcn_wmma_f32_16x16x32_bf16(false, a, false, b1, (short)0, az,  false, false);
        anh = __builtin_amdgcn_wmma_f32_16x16x32_bf16(false, a, false, b2, (short)0, anh, false, false);
    }

    // gate math; C/D layout: n = lane&15, m = vgpr + 8*(lane>>4)
    const float*  hf_in  = hf_base   + off_in;
    float*        hf_out = hf_base_w + off_out;
    __bf16*       hb_out = hb_base_w + off_out;
    #pragma unroll
    for (int i = 0; i < 8; ++i) {
        const int m = wave * 16 + i + 8 * khalf;
        float r  = sigmoidf_(ar[i] + br);
        float z  = sigmoidf_(az[i] + bz);
        float nn = tanhf_(ani[i] + bni + r * (anh[i] + bnh));
        float hp = hf_in[(size_t)m * L_ + col];
        float hn = (1.0f - z) * nn + z * hp;
        hf_out[(size_t)m * L_ + col] = hn;
        hb_out[(size_t)m * L_ + col] = (__bf16)hn;
    }
}

// Output projection (device): x = h @ W_out.T + b_out, W_out from LDS.
// Only blocks 0..7 call this (nt = blk).
__device__ __forceinline__ void proj_dev(
    const char* smem, const __bf16* __restrict__ hb_base, size_t off_in,
    float bo, float* __restrict__ out_f32, int s_store, bool write_out,
    __bf16* __restrict__ xc_base, size_t xc_off,
    int nt, int lane, int wave)
{
    const int khalf = lane >> 4;
    const int mloc  = lane & 15;
    const int arow  = wave * 16 + mloc;
    const int f     = nt * 16 + mloc;

    v8f acc = {};
    const __bf16* hrow = hb_base + off_in + (size_t)arow * L_;
    #pragma unroll 2
    for (int kb = 0; kb < KBO; ++kb) {
        v16bf a = load_a_frag(hrow, kb * 32, khalf);
        v16bf b = lds_frag(smem, WOUT_OFF + ((kb * 32 + lane) << 5));
        acc = __builtin_amdgcn_wmma_f32_16x16x32_bf16(false, a, false, b, (short)0, acc, false, false);
    }

    __bf16* x_next = xc_base + xc_off;
    #pragma unroll
    for (int i = 0; i < 8; ++i) {
        const int b = wave * 16 + i + 8 * khalf;
        const float v = acc[i] + bo;
        if (write_out)
            out_f32[(size_t)b * S_ * F_ + (size_t)s_store * F_ + f] = v;
        x_next[(size_t)b * F_ + f] = (__bf16)v;
    }
}

// ---------------------------------------------------------------------------
// Persistent kernel: the entire encoder + decoder recurrence.
// ---------------------------------------------------------------------------
__global__ __launch_bounds__(256) void rae_persistent_kernel(
    const __bf16* __restrict__ xT,
    const __bf16* __restrict__ WihE_p, const __bf16* __restrict__ WhhE_p,
    const __bf16* __restrict__ WihD_p, const __bf16* __restrict__ WhhD_p,
    const __bf16* __restrict__ Wout_p,
    const float* __restrict__ bihE, const float* __restrict__ bhhE,
    const float* __restrict__ bihD, const float* __restrict__ bhhD,
    const float* __restrict__ bout,
    float* __restrict__ hf,      // 2 * B * L f32   (ping-pong, contiguous)
    __bf16* __restrict__ hb,     // 2 * B * L bf16  (ping-pong, contiguous)
    __bf16* __restrict__ xc,     // 2 * B * F bf16  (ping-pong, contiguous)
    float* __restrict__ out, unsigned* __restrict__ bar)
{
    __shared__ __align__(16) char smem[SMEM_BYTES];
    const int blk  = blockIdx.x;           // latent column tile 0..63
    const int lane = threadIdx.x & 31;
    const int wave = threadIdx.x >> 5;
    const int mloc = lane & 15;
    const int col  = blk * 16 + mloc;

    // ---- stage encoder weights (+ W_out for proj blocks) into LDS ----
    #pragma unroll
    for (int g = 0; g < 3; ++g) {
        copy_to_lds(smem, WHH_OFF + g * (KBH * 32 * 32),
                    WhhE_p + (size_t)(g * NTG + blk) * KBH * 512, KBH * 32 * 32);
        copy_to_lds(smem, WIH_OFF + g * (KBI * 32 * 32),
                    WihE_p + (size_t)(g * NTG + blk) * KBI * 512, KBI * 32 * 32);
    }
    if (blk < F_ / 16)
        copy_to_lds(smem, WOUT_OFF, Wout_p + (size_t)blk * KBO * 512, KBO * 32 * 32);
    __syncthreads();

    // hoisted per-column biases
    const float brE  = bihE[col] + bhhE[col];
    const float bzE  = bihE[L_ + col] + bhhE[L_ + col];
    const float bniE = bihE[2 * L_ + col];
    const float bnhE = bhhE[2 * L_ + col];
    const float bo   = (blk < F_ / 16) ? bout[blk * 16 + mloc] : 0.0f;

    unsigned epoch = 0;
    int cur = 0;

    // ---- encoder scan: 512 sequential GRU steps ----
    for (int t = 0; t < S_; ++t) {
        const size_t oin  = (size_t)cur * HSTRIDE;
        const size_t oout = (size_t)(cur ^ 1) * HSTRIDE;
        gru_step_dev(smem, hb, hf, oin, oout, xT + (size_t)t * B_ * F_,
                     brE, bzE, bniE, bnhE, hf, hb, blk, lane, wave);
        cur ^= 1;
        grid_barrier(bar, ++epoch);
    }

    // ---- x_init = h_n @ W_out.T + b_out ; swap LDS to decoder weights ----
    if (blk < F_ / 16)
        proj_dev(smem, hb, (size_t)cur * HSTRIDE, bo, out, 0, false,
                 xc, 0, blk, lane, wave);
    #pragma unroll
    for (int g = 0; g < 3; ++g) {
        copy_to_lds(smem, WHH_OFF + g * (KBH * 32 * 32),
                    WhhD_p + (size_t)(g * NTG + blk) * KBH * 512, KBH * 32 * 32);
        copy_to_lds(smem, WIH_OFF + g * (KBI * 32 * 32),
                    WihD_p + (size_t)(g * NTG + blk) * KBI * 512, KBI * 32 * 32);
    }
    __syncthreads();
    grid_barrier(bar, ++epoch);

    const float brD  = bihD[col] + bhhD[col];
    const float bzD  = bihD[L_ + col] + bhhD[L_ + col];
    const float bniD = bihD[2 * L_ + col];
    const float bnhD = bhhD[2 * L_ + col];

    // ---- decoder: 512 sequential GRUCell + Linear steps, output flipped ----
    for (int t = 0; t < S_; ++t) {
        const size_t oin  = (size_t)cur * HSTRIDE;
        const size_t oout = (size_t)(cur ^ 1) * HSTRIDE;
        gru_step_dev(smem, hb, hf, oin, oout, xc + (size_t)(t & 1) * XSTRIDE,
                     brD, bzD, bniD, bnhD, hf, hb, blk, lane, wave);
        cur ^= 1;
        grid_barrier(bar, ++epoch);
        if (blk < F_ / 16)
            proj_dev(smem, hb, (size_t)cur * HSTRIDE, bo, out, S_ - 1 - t, true,
                     xc, (size_t)((t + 1) & 1) * XSTRIDE, blk, lane, wave);
        grid_barrier(bar, ++epoch);
    }
}

// ---------------------------------------------------------------------------
// Host-side orchestration (all on `stream`, graph-capture safe).
// ---------------------------------------------------------------------------
extern "C" void kernel_launch(void* const* d_in, const int* in_sizes, int n_in,
                              void* d_out, int out_size, void* d_ws, size_t ws_size,
                              hipStream_t stream) {
    (void)in_sizes; (void)n_in; (void)out_size; (void)ws_size;

    const float* x      = (const float*)d_in[0];
    const float* W_ih_e = (const float*)d_in[1];
    const float* W_hh_e = (const float*)d_in[2];
    const float* b_ih_e = (const float*)d_in[3];
    const float* b_hh_e = (const float*)d_in[4];
    const float* W_ih_d = (const float*)d_in[5];
    const float* W_hh_d = (const float*)d_in[6];
    const float* b_ih_d = (const float*)d_in[7];
    const float* b_hh_d = (const float*)d_in[8];
    const float* W_out  = (const float*)d_in[9];
    const float* b_out  = (const float*)d_in[10];
    float* out = (float*)d_out;

    // ---- workspace layout (256B aligned) ----
    char* ws = (char*)d_ws;
    size_t off = 0;
    auto take = [&](size_t bytes) -> char* {
        char* p = ws + off;
        off = (off + bytes + 255) & ~(size_t)255;
        return p;
    };
    __bf16*   xT      = (__bf16*)take((size_t)B_ * S_ * F_ * 2);
    __bf16*   Wih_e_p = (__bf16*)take((size_t)3 * L_ * F_ * 2);
    __bf16*   Whh_e_p = (__bf16*)take((size_t)3 * L_ * L_ * 2);
    __bf16*   Wih_d_p = (__bf16*)take((size_t)3 * L_ * F_ * 2);
    __bf16*   Whh_d_p = (__bf16*)take((size_t)3 * L_ * L_ * 2);
    __bf16*   Wout_p  = (__bf16*)take((size_t)F_ * L_ * 2);
    float*    hf      = (float*)take(2 * HSTRIDE * 4);
    __bf16*   hb      = (__bf16*)take(2 * HSTRIDE * 2);
    __bf16*   xc      = (__bf16*)take(2 * XSTRIDE * 2);
    unsigned* bar     = (unsigned*)take(256);

    // ---- one-time prep ----
    {
        int n = B_ * S_ * F_;
        xconv_kernel<<<(n + 255) / 256, 256, 0, stream>>>(x, xT);
    }
    auto pack = [&](const float* W, __bf16* dst, int N, int K) {
        int n = (N * K) >> 1;
        prepack_kernel<<<(n + 255) / 256, 256, 0, stream>>>(W, dst, N, K);
    };
    pack(W_ih_e, Wih_e_p, 3 * L_, F_);
    pack(W_hh_e, Whh_e_p, 3 * L_, L_);
    pack(W_ih_d, Wih_d_p, 3 * L_, F_);
    pack(W_hh_d, Whh_d_p, 3 * L_, L_);
    pack(W_out,  Wout_p,  F_,     L_);
    {
        int n = B_ * L_;
        init_kernel<<<(n + 255) / 256, 256, 0, stream>>>(hf, hb, bar);
    }

    // ---- the whole recurrence in one persistent kernel ----
    rae_persistent_kernel<<<GRID_, 256, 0, stream>>>(
        xT, Wih_e_p, Whh_e_p, Wih_d_p, Whh_d_p, Wout_p,
        b_ih_e, b_hh_e, b_ih_d, b_hh_d, b_out,
        hf, hb, xc, out, bar);
}
